// IBPathBlock_85650237817327
// MI455X (gfx1250) — matrix operand order repaired and checked by reading, not compiled, not run on hardware
//
#include <hip/hip_runtime.h>
#include <hip/hip_bf16.h>

// ---------------------------------------------------------------------------
// Fused survival-VAE forward for MI455X (gfx1250), wave32 + WMMA bf16 +
// async global->LDS weight staging (double-buffered) + eps prefetch.
//
//   h   = relu(x @ W_enc^T + b_enc)
//   mu  = h @ W_mu^T + b_mu
//   std = softplus(h @ W_std^T + b_std - 5) + 1e-6
//   z   = mu + std*eps ; pred = z @ W_dec^T + b_dec (K=4, N padded to 16)
//   out[i] = nll_surv(pred_i, y_i, c_i) + 0.5*mean(mu^2+std^2-2*log(std)-1)
//
// Roofline: ~103 GFLOP vs ~0.51 GB min HBM traffic (x + eps + out) -> at the
// compute/memory ridge; weights live in L2 (pre-converted to bf16 in d_ws and
// DMA'd to LDS with GLOBAL_LOAD_ASYNC_TO_LDS_B128 so staging overlaps WMMA).
// ---------------------------------------------------------------------------

typedef __attribute__((ext_vector_type(16))) __bf16 v16bf;
typedef __attribute__((ext_vector_type(8)))  __bf16 v8bf;
typedef __attribute__((ext_vector_type(8)))  float  v8f;

#define BB  262144
#define DD  256
#define KK  4
#define ROWS_PER_BLOCK 128
#define NBLOCKS (BB / ROWS_PER_BLOCK)     // 2048
#define NTILES  16                        // 256 / 16 (N)
#define KTILES  8                         // 256 / 32 (K per wmma)

// Each weight matrix: 256 rows x 256 bf16, XOR-swizzled on 32B chunks:
//   element (n,k) -> byte n*512 + (((k>>4) ^ (n&15)) << 5) + (k&15)*2
#define WBYTES  (DD * DD * 2)             // 131072
#define WCHUNKS (WBYTES / 16)             // 8192 16B chunks
#define WITERS  (WCHUNKS / 256)           // 32 async b128 per thread / per wave

// LDS: two weight buffers (double-buffered DMA) + per-wave h/z tile.
// 2*131072 + 8*8192 = 327680 B = the full 320KB WGP LDS.
#define LDS_W0_OFF 0
#define LDS_W1_OFF WBYTES
#define LDS_H_OFF  (2 * WBYTES)
#define HWAVE_BYTES 8192                  // 16 rows * 512B, swizzled like W
#define LDS_TOTAL  (LDS_H_OFF + 8 * HWAVE_BYTES)

// d_ws layout (bytes)
#define WS_SCALAR   0                     // 1 float
#define WS_PARTIALS 64                    // NBLOCKS floats
#define WS_WENC     16384
#define WS_WMU      (WS_WENC + WBYTES)
#define WS_WSTD     (WS_WMU + WBYTES)
#define WS_WDEC     (WS_WSTD + WBYTES)    // 16x256 bf16, rows 4..15 zero
#define WS_END      (WS_WDEC + 16 * DD * 2)

// ---------------- async / wait helpers ----------------
typedef int v4i __attribute__((vector_size(16)));
typedef __attribute__((address_space(3))) v4i lds_v4i;
typedef __attribute__((address_space(3))) char las_char;

#if __has_builtin(__builtin_amdgcn_s_wait_asynccnt)
#define WAIT_ASYNC(n) __builtin_amdgcn_s_wait_asynccnt(n)
#else
#define WAIT_ASYNC(n) asm volatile("s_wait_asynccnt %0" ::"i"(n) : "memory")
#endif

static __device__ inline void async_b128(const void* g, void* l) {
#if __has_builtin(__builtin_amdgcn_global_load_async_to_lds_b128)
  __builtin_amdgcn_global_load_async_to_lds_b128(
      (v4i*)g, (lds_v4i*)l, 0, 0);
#else
  unsigned lo = (unsigned)(uintptr_t)(las_char*)l;
  unsigned long long ga = (unsigned long long)(uintptr_t)g;
  asm volatile("global_load_async_to_lds_b128 %0, %1, off"
               :: "v"(lo), "v"(ga) : "memory");
#endif
}

// Stage one 128KB bf16 weight matrix (already swizzled in d_ws) into LDS.
// Exactly WITERS(=32) async instructions per wave.
static __device__ inline void stage_weight(const __bf16* __restrict__ gsrc,
                                           __bf16* ldst, int tid) {
#pragma unroll
  for (int it = 0; it < WITERS; ++it) {
    int e = (it * 256 + tid) * 8;          // 8 bf16 = 16B per lane
    async_b128(gsrc + e, ldst + e);
  }
}

// ---------------- fragment helpers ----------------
static __device__ inline v8f wmma_bf16(v16bf a, v16bf b, v8f c) {
  return __builtin_amdgcn_wmma_f32_16x16x32_bf16(
      false, a, false, b, (short)0, c, false, false);
}

// B-fragment (32x16, K x N) from swizzled LDS weight buffer:
// lane column n, 16 consecutive k starting at kcol (multiple of 16).
static __device__ inline v16bf load_bfrag(const __bf16* lw, int n, int kcol) {
  int p = (kcol >> 4) ^ (n & 15);
  return *reinterpret_cast<const v16bf*>(lw + (n << 8) + (p << 4));
}

// A-fragment (16x32) from a global fp32 row: frag[0..15]=k0..k0+15
static __device__ inline v16bf afrag_from_f32(const float* rowp, int k0) {
  const float4 f0 = *reinterpret_cast<const float4*>(rowp + k0);
  const float4 f1 = *reinterpret_cast<const float4*>(rowp + k0 + 4);
  const float4 f2 = *reinterpret_cast<const float4*>(rowp + k0 + 8);
  const float4 f3 = *reinterpret_cast<const float4*>(rowp + k0 + 12);
  v16bf a;
  a[0]=(__bf16)f0.x;  a[1]=(__bf16)f0.y;  a[2]=(__bf16)f0.z;  a[3]=(__bf16)f0.w;
  a[4]=(__bf16)f1.x;  a[5]=(__bf16)f1.y;  a[6]=(__bf16)f1.z;  a[7]=(__bf16)f1.w;
  a[8]=(__bf16)f2.x;  a[9]=(__bf16)f2.y;  a[10]=(__bf16)f2.z; a[11]=(__bf16)f2.w;
  a[12]=(__bf16)f3.x; a[13]=(__bf16)f3.y; a[14]=(__bf16)f3.z; a[15]=(__bf16)f3.w;
  return a;
}

// A-fragment from the wave's swizzled LDS h/z tile (row m, k0 in {kt*32, +16}).
static __device__ inline v16bf afrag_from_lds(const __bf16* hbase, int m, int k0) {
  int p0 = ((k0 >> 4)        ^ (m & 15)) << 4;
  v8bf lo = *reinterpret_cast<const v8bf*>(hbase + (m << 8) + p0);
  v8bf hi = *reinterpret_cast<const v8bf*>(hbase + (m << 8) + p0 + 8);
  v16bf a;
#pragma unroll
  for (int j = 0; j < 8; ++j) { a[j] = lo[j]; a[8 + j] = hi[j]; }
  return a;
}

static __device__ inline void store_h(__bf16* hbase, int m, int n, float v) {
  int p = ((n >> 4) ^ (m & 15)) << 4;
  hbase[(m << 8) + p + (n & 15)] = (__bf16)v;
}

static __device__ inline float lclip(float v) { return __logf(fmaxf(v, 1e-7f)); }

// ---------------- prep kernels: fp32 -> swizzled bf16 in d_ws ----------------
__global__ void __launch_bounds__(256)
prep_weight(const float* __restrict__ W, __bf16* __restrict__ out) {
  int i = blockIdx.x * 256 + threadIdx.x;   // over 256*256 elements
  int n = i >> 8, k = i & 255;
  int p = ((k >> 4) ^ (n & 15)) << 4;
  out[(n << 8) + p + (k & 15)] = (__bf16)W[i];
}

__global__ void __launch_bounds__(256)
prep_wdec(const float* __restrict__ Wd, __bf16* __restrict__ out) {
  int i = blockIdx.x * 256 + threadIdx.x;   // over 16*256 elements (unswizzled)
  int n = i >> 8, k = i & 255;
  out[i] = (n < KK) ? (__bf16)Wd[n * DD + k] : (__bf16)0.f;
}

// ---------------- main fused kernel ----------------
__global__ void __launch_bounds__(256)
fused_vae_surv(const float* __restrict__ x,
               const float* __restrict__ censor,
               const float* __restrict__ eps,
               const int*   __restrict__ y_disc,
               const __bf16* __restrict__ wenc_g,
               const __bf16* __restrict__ wmu_g,
               const __bf16* __restrict__ wstd_g,
               const __bf16* __restrict__ wdec_g,
               const float* __restrict__ b_enc, const float* __restrict__ b_mu,
               const float* __restrict__ b_std, const float* __restrict__ b_dec,
               float* __restrict__ out, float* __restrict__ kl_partials) {
  extern __shared__ __align__(32) char smem[];
  __bf16* bufW0 = (__bf16*)(smem + LDS_W0_OFF);
  __bf16* bufW1 = (__bf16*)(smem + LDS_W1_OFF);

  const int tid   = threadIdx.x;
  const int lane  = tid & 31;
  const int wave  = tid >> 5;
  const int ln16  = lane & 15;
  const int lhi   = lane >> 4;
  const int khalf = lhi * 16;
  const int mbase = lhi * 8;

  const int wave_row0 = blockIdx.x * ROWS_PER_BLOCK + wave * 16;
  __bf16* hwave = (__bf16*)(smem + LDS_H_OFF + wave * HWAVE_BYTES);

  // DMA W_enc into buf0 while we fetch this wave's x rows + prefetch eps.
  stage_weight(wenc_g, bufW0, tid);

  const float* xrow = x + (size_t)(wave_row0 + ln16) * DD;
  v16bf ax[KTILES];
#pragma unroll
  for (int kt = 0; kt < KTILES; ++kt)
    ax[kt] = afrag_from_f32(xrow, kt * 32 + khalf);
#pragma unroll
  for (int r = 0; r < 16; ++r)   // pull eps tile toward L2/L0 for the gather
    __builtin_prefetch(eps + (size_t)(wave_row0 + r) * DD + lane * 8, 0, 3);

  WAIT_ASYNC(0);
  __syncthreads();                       // bufW0 (W_enc) ready for all waves

  // ---------------- Layer 1: h = relu(x W_enc^T + b_enc) ----------------
  stage_weight(wmu_g, bufW1, tid);       // overlap W_mu DMA with layer-1 WMMAs
#pragma unroll
  for (int t = 0; t < NTILES; ++t) {
    v8f acc = {};
#pragma unroll
    for (int kt = 0; kt < KTILES; ++kt)
      acc = wmma_bf16(ax[kt], load_bfrag(bufW0, t * 16 + ln16, kt * 32 + lhi * 16), acc);
    const int n = t * 16 + ln16;
    const float bias = b_enc[n];
#pragma unroll
    for (int r = 0; r < 8; ++r) {
      float v = acc[r] + bias;
      store_h(hwave, mbase + r, n, v > 0.f ? v : 0.f);
    }
  }
  __syncthreads();                       // all waves done reading bufW0

  stage_weight(wstd_g, bufW0, tid);      // overlap W_std DMA with layer-2 WMMAs
  WAIT_ASYNC(WITERS);                    // W_mu (first 32, in-order) retired
  __syncthreads();                       // bufW1 (W_mu) ready for all waves

  const __bf16* hrow = hwave;            // wave-local tile; lane row = ln16
  v16bf ah[KTILES];
#pragma unroll
  for (int kt = 0; kt < KTILES; ++kt)
    ah[kt] = afrag_from_lds(hrow, ln16, kt * 32 + khalf);

  // ---------------- Layer 2: mu ----------------
  v8f mu_acc[NTILES];
#pragma unroll
  for (int t = 0; t < NTILES; ++t) {
    v8f acc = {};
#pragma unroll
    for (int kt = 0; kt < KTILES; ++kt)
      acc = wmma_bf16(ah[kt], load_bfrag(bufW1, t * 16 + ln16, kt * 32 + lhi * 16), acc);
    mu_acc[t] = acc;
  }
  WAIT_ASYNC(0);
  __syncthreads();                       // bufW0 (W_std) ready for all waves

  // ---------------- Layer 3: std ----------------
  v8f sd_acc[NTILES];
#pragma unroll
  for (int t = 0; t < NTILES; ++t) {
    v8f acc = {};
#pragma unroll
    for (int kt = 0; kt < KTILES; ++kt)
      acc = wmma_bf16(ah[kt], load_bfrag(bufW0, t * 16 + ln16, kt * 32 + lhi * 16), acc);
    sd_acc[t] = acc;
  }

  // ---------------- z = mu + std*eps ; KL partial ; z -> own LDS tile ------
  float kl_local = 0.f;
#pragma unroll
  for (int t = 0; t < NTILES; ++t) {
    const int n = t * 16 + ln16;
    const float bmu = b_mu[n];
    const float bsd = b_std[n];
#pragma unroll
    for (int r = 0; r < 8; ++r) {
      const int m = mbase + r;
      float muv = mu_acc[t][r] + bmu;
      float sv  = sd_acc[t][r] + bsd - 5.0f;
      float sp  = (sv > 20.f) ? sv : __logf(1.f + __expf(sv));
      float stdv = sp + 1e-6f;
      float e = eps[(size_t)(wave_row0 + m) * DD + n];
      kl_local += muv * muv + stdv * stdv - 2.f * __logf(stdv) - 1.f;
      store_h(hwave, m, n, muv + stdv * e);
    }
  }

  // ---------------- pred = z W_dec^T + b_dec (bf16 W_dec pre-padded) -------
  v16bf az[KTILES];
#pragma unroll
  for (int kt = 0; kt < KTILES; ++kt)    // read ALL z frags before pred write
    az[kt] = afrag_from_lds(hwave, ln16, kt * 32 + khalf);
  v8f pacc = {};
#pragma unroll
  for (int kt = 0; kt < KTILES; ++kt) {
    v16bf b = *reinterpret_cast<const v16bf*>(
        wdec_g + (ln16 << 8) + kt * 32 + lhi * 16);
    pacc = wmma_bf16(az[kt], b, pacc);
  }
  float* pp = (float*)hwave;             // reuse own tile (z consumed): 16x16 f32
  const float bd = (ln16 < KK) ? b_dec[ln16] : 0.f;
#pragma unroll
  for (int r = 0; r < 8; ++r)
    pp[(mbase + r) * 16 + ln16] = pacc[r] + bd;

  // ---------------- per-row survival NLL (lanes 0..15) -------------------
  if (lane < 16) {
    const int m = lane;
    const int rowg = wave_row0 + m;
    const float* pr = pp + m * 16;
    float h0 = 1.f / (1.f + __expf(-pr[0]));
    float h1 = 1.f / (1.f + __expf(-pr[1]));
    float h2 = 1.f / (1.f + __expf(-pr[2]));
    float h3 = 1.f / (1.f + __expf(-pr[3]));
    float S1 = 1.f - h0;
    float S2 = S1 * (1.f - h1);
    float S3 = S2 * (1.f - h2);
    float S4 = S3 * (1.f - h3);
    int   y = y_disc[rowg];
    float c = censor[rowg];
    float s_prev = (y == 0) ? 1.f : (y == 1) ? S1 : (y == 2) ? S2 : S3;
    float h_y    = (y == 0) ? h0  : (y == 1) ? h1 : (y == 2) ? h2 : h3;
    float s_cur  = (y == 0) ? S1  : (y == 1) ? S2 : (y == 2) ? S3 : S4;
    out[rowg] = -(1.f - c) * (lclip(s_prev) + lclip(h_y)) - c * lclip(s_cur);
  }

  // ---------------- block KL partial ----------------
#pragma unroll
  for (int off = 16; off > 0; off >>= 1)
    kl_local += __shfl_xor(kl_local, off, 32);
  if (lane == 0) pp[256] = kl_local;     // own tile, past the 16x16 pred area
  __syncthreads();
  if (tid == 0) {
    float s = 0.f;
#pragma unroll
    for (int w = 0; w < 8; ++w)
      s += ((float*)(smem + LDS_H_OFF + w * HWAVE_BYTES))[256];
    kl_partials[blockIdx.x] = s;
  }
}

__global__ void __launch_bounds__(256)
kl_reduce(const float* __restrict__ partials, float* __restrict__ scalar, int n) {
  __shared__ float sm[256];
  float s = 0.f;
  for (int i = threadIdx.x; i < n; i += 256) s += partials[i];
  sm[threadIdx.x] = s;
  __syncthreads();
  for (int st = 128; st > 0; st >>= 1) {
    if (threadIdx.x < st) sm[threadIdx.x] += sm[threadIdx.x + st];
    __syncthreads();
  }
  if (threadIdx.x == 0)
    scalar[0] = 0.5f * sm[0] / ((float)BB * (float)DD);
}

__global__ void __launch_bounds__(256)
add_kl(float* __restrict__ out, const float* __restrict__ scalar) {
  int i = blockIdx.x * blockDim.x + threadIdx.x;
  out[i] += scalar[0];
}

extern "C" void kernel_launch(void* const* d_in, const int* in_sizes, int n_in,
                              void* d_out, int out_size, void* d_ws, size_t ws_size,
                              hipStream_t stream) {
  (void)in_sizes; (void)n_in; (void)out_size; (void)ws_size;
  const float* x      = (const float*)d_in[0];
  // d_in[1] = event_time (unused by the reference computation)
  const float* censor = (const float*)d_in[2];
  const float* eps    = (const float*)d_in[3];
  const int*   y_disc = (const int*)  d_in[4];
  const float* W_enc  = (const float*)d_in[5];
  const float* b_enc  = (const float*)d_in[6];
  const float* W_mu   = (const float*)d_in[7];
  const float* b_mu   = (const float*)d_in[8];
  const float* W_std  = (const float*)d_in[9];
  const float* b_std  = (const float*)d_in[10];
  const float* W_dec  = (const float*)d_in[11];
  const float* b_dec  = (const float*)d_in[12];

  float* out = (float*)d_out;
  char*  ws  = (char*)d_ws;
  float* kl_scalar   = (float*)(ws + WS_SCALAR);
  float* kl_partials = (float*)(ws + WS_PARTIALS);
  __bf16* wenc_b = (__bf16*)(ws + WS_WENC);
  __bf16* wmu_b  = (__bf16*)(ws + WS_WMU);
  __bf16* wstd_b = (__bf16*)(ws + WS_WSTD);
  __bf16* wdec_b = (__bf16*)(ws + WS_WDEC);

  // Pre-convert weights to (swizzled) bf16 once per launch — tiny cost.
  prep_weight<<<DD * DD / 256, 256, 0, stream>>>(W_enc, wenc_b);
  prep_weight<<<DD * DD / 256, 256, 0, stream>>>(W_mu,  wmu_b);
  prep_weight<<<DD * DD / 256, 256, 0, stream>>>(W_std, wstd_b);
  prep_wdec  <<<16 * DD / 256, 256, 0, stream>>>(W_dec, wdec_b);

  fused_vae_surv<<<NBLOCKS, 256, LDS_TOTAL, stream>>>(
      x, censor, eps, y_disc, wenc_b, wmu_b, wstd_b, wdec_b,
      b_enc, b_mu, b_std, b_dec, out, kl_partials);
  kl_reduce<<<1, 256, 0, stream>>>(kl_partials, kl_scalar, NBLOCKS);
  add_kl<<<BB / 256, 256, 0, stream>>>(out, kl_scalar);
}
